// Attention_60567628808865
// MI455X (gfx1250) — matrix operand order repaired and checked by reading, not compiled, not run on hardware
//
#include <hip/hip_runtime.h>
#include <hip/hip_bf16.h>
#include <math.h>

// ---------------------------------------------------------------------------
// Problem constants: B=4, S=1024, H=4096, NH=32, D=128
// ---------------------------------------------------------------------------
typedef __bf16 bf16_t;
typedef __attribute__((ext_vector_type(16))) __bf16 v16bf;
typedef __attribute__((ext_vector_type(8)))  float   v8f;

#define WMMA_BF16(a, b, c) \
  __builtin_amdgcn_wmma_f32_16x16x32_bf16(false, (a), false, (b), (short)0, (c), false, false)

// ---- CDNA5 async global->LDS copy (no VGPR data path, tracked on ASYNCcnt) -
// LDS address comes from a VGPR; generic pointers to LDS carry the LDS offset
// in their low 32 bits (aperture bits live in [63:32]).
__device__ __forceinline__ void async_copy_b128(void* lds_dst, const void* gsrc) {
  const unsigned lds = (unsigned)(unsigned long long)lds_dst;
  const unsigned long long ga = (unsigned long long)gsrc;
  asm volatile("global_load_async_to_lds_b128 %0, %1, off"
               :: "v"(lds), "v"(ga) : "memory");
}
__device__ __forceinline__ void wait_async() {
  asm volatile("s_wait_asynccnt 0x0" ::: "memory");
}

// ---- WMMA fragment loaders (row-major source, ld in elements) -------------
// A fragment 16(M)x32(K): lane L holds row M=L%16; hh=L/16:
//   element e<8  -> K = 8*hh + e
//   element e>=8 -> K = 16 + 8*hh + (e-8)          (ISA 7.12.2, 16-bit A)
__device__ __forceinline__ v16bf load_a_frag(const bf16_t* p, int row0, int k0, int ld) {
  const int lane = threadIdx.x & 31;
  const int hh = lane >> 4;
  const bf16_t* r = p + (size_t)(row0 + (lane & 15)) * ld + k0;
  v16bf a;
#pragma unroll
  for (int e = 0; e < 8; ++e) a[e] = r[8 * hh + e];
#pragma unroll
  for (int e = 0; e < 8; ++e) a[8 + e] = r[16 + 8 * hh + e];
  return a;
}

// B fragment 32(K)x16(N): lane L holds N = L%16; hh=L/16:
//   element e<8 -> K = 8*hh + e; element e>=8 -> K = 16 + 8*hh + (e-8)
__device__ __forceinline__ v16bf load_b_frag(const bf16_t* p, int k0, int col0, int ld) {
  const int lane = threadIdx.x & 31;
  const int hh = lane >> 4;
  const bf16_t* r = p + (size_t)k0 * ld + col0 + (lane & 15);
  v16bf b;
#pragma unroll
  for (int e = 0; e < 8; ++e) b[e] = r[(size_t)(8 * hh + e) * ld];
#pragma unroll
  for (int e = 0; e < 8; ++e) b[8 + e] = r[(size_t)(16 + 8 * hh + e) * ld];
  return b;
}

// 16-lane-half reductions (wave32; xor masks < 16 keep lanes in their half)
__device__ __forceinline__ float half_max(float v) {
  v = fmaxf(v, __shfl_xor(v, 1, 32));
  v = fmaxf(v, __shfl_xor(v, 2, 32));
  v = fmaxf(v, __shfl_xor(v, 4, 32));
  v = fmaxf(v, __shfl_xor(v, 8, 32));
  return v;
}
__device__ __forceinline__ float half_sum(float v) {
  v += __shfl_xor(v, 1, 32);
  v += __shfl_xor(v, 2, 32);
  v += __shfl_xor(v, 4, 32);
  v += __shfl_xor(v, 8, 32);
  return v;
}

// ---------------------------------------------------------------------------
// Kernel 1: QKV projection.  X[4096,4096]f32 @ W_pack[4096,12288]f32 -> bf16
// Q/K/V buffers in [B,NH,S,D] layout.  128x128 tile, 8 waves (2Mx4N),
// each wave 64x32 = 8 accumulators, K-step 32 (one WMMA depth).
// (f32->bf16 conversion needed, so both tiles go through VGPRs.)
// ---------------------------------------------------------------------------
__global__ __launch_bounds__(256) void qkv_gemm_kernel(
    const float* __restrict__ X, const float* __restrict__ W,
    bf16_t* __restrict__ Qb, bf16_t* __restrict__ Kb, bf16_t* __restrict__ Vb) {
  constexpr int LDA = 48;    // 32 + pad (96B rows: 16B aligned)
  constexpr int LDB = 136;   // 128 + pad
  __shared__ __align__(16) bf16_t As[128 * LDA];
  __shared__ __align__(16) bf16_t Bs[32 * LDB];
  const int tid = threadIdx.x;
  const int lane = tid & 31, w = tid >> 5;
  const int wm = w & 1, wn = w >> 1;
  const int m0 = blockIdx.y * 128, n0 = blockIdx.x * 128;

  v8f acc[4][2];
#pragma unroll
  for (int i = 0; i < 4; ++i)
#pragma unroll
    for (int j = 0; j < 2; ++j) acc[i][j] = {};

  for (int kt = 0; kt < 128; ++kt) {
    const int k0 = kt * 32;
    __syncthreads();
    {  // A tile 128x32, f32 -> bf16
      const int r = tid >> 3, c = (tid & 7) * 4;
#pragma unroll
      for (int i = 0; i < 4; ++i) {
        const float4 v = *(const float4*)(X + (size_t)(m0 + r + i * 32) * 4096 + k0 + c);
        bf16_t* d = &As[(r + i * 32) * LDA + c];
        d[0] = (bf16_t)v.x; d[1] = (bf16_t)v.y; d[2] = (bf16_t)v.z; d[3] = (bf16_t)v.w;
      }
    }
    {  // B tile 32x128, f32 -> bf16
      const int r = tid >> 5, c = (tid & 31) * 4;
#pragma unroll
      for (int i = 0; i < 4; ++i) {
        const float4 v = *(const float4*)(W + (size_t)(k0 + r + i * 8) * 12288 + n0 + c);
        bf16_t* d = &Bs[(r + i * 8) * LDB + c];
        d[0] = (bf16_t)v.x; d[1] = (bf16_t)v.y; d[2] = (bf16_t)v.z; d[3] = (bf16_t)v.w;
      }
    }
    if (kt + 1 < 128) {  // prefetch next tiles -> global_prefetch_b8
      __builtin_prefetch(X + (size_t)(m0 + (tid >> 3)) * 4096 + k0 + 32, 0, 1);
      __builtin_prefetch(W + (size_t)(k0 + 32 + (tid >> 5)) * 12288 + n0, 0, 1);
    }
    __syncthreads();

    v16bf af[4], bfr[2];
#pragma unroll
    for (int mi = 0; mi < 4; ++mi) af[mi] = load_a_frag(As, wm * 64 + mi * 16, 0, LDA);
#pragma unroll
    for (int ni = 0; ni < 2; ++ni) bfr[ni] = load_b_frag(Bs, 0, wn * 32 + ni * 16, LDB);
#pragma unroll
    for (int mi = 0; mi < 4; ++mi)
#pragma unroll
      for (int ni = 0; ni < 2; ++ni) acc[mi][ni] = WMMA_BF16(af[mi], bfr[ni], acc[mi][ni]);
  }

  // Epilogue: n0 is 128-aligned so the whole WG tile sits in one of Q/K/V.
  const int part = n0 >> 12;
  bf16_t* dst = (part == 0) ? Qb : ((part == 1) ? Kb : Vb);
  const int half = lane >> 4, cn = lane & 15;
#pragma unroll
  for (int mi = 0; mi < 4; ++mi)
#pragma unroll
    for (int ni = 0; ni < 2; ++ni)
#pragma unroll
      for (int r = 0; r < 8; ++r) {
        const int gm = m0 + wm * 64 + mi * 16 + r + 8 * half;  // row in [0,4096)
        const int gn = n0 + wn * 32 + ni * 16 + cn;            // col in [0,12288)
        const int hcol = gn & 4095, head = hcol >> 7, d = hcol & 127;
        const int b = gm >> 10, s = gm & 1023;
        dst[((size_t)(b * 32 + head) * 1024 + s) * 128 + d] = (bf16_t)acc[mi][ni][r];
      }
}

// ---------------------------------------------------------------------------
// Kernel 2: in-place RoPE on Q and K.  One thread per (b,h,s,j) pair, j<64,
// rotates (x_j, x_{j+64}).  position_ids == arange(S) in the reference.
// ---------------------------------------------------------------------------
__global__ __launch_bounds__(256) void rope_kernel(bf16_t* __restrict__ Qb,
                                                   bf16_t* __restrict__ Kb) {
  const int idx = blockIdx.x * 256 + threadIdx.x;  // < 4*32*1024*64
  const int j = idx & 63;
  const int s = (idx >> 6) & 1023;
  const int bh = idx >> 16;
  const size_t base = ((size_t)bh * 1024 + s) * 128;
  // inv_freq = 10000^(-j/64) = 2^(-j * log2(10000)/64)
  const float inv = exp2f(-(float)j * (13.287712379549449f / 64.0f));
  const float ang = (float)s * inv;
  const float c = cosf(ang), sn = sinf(ang);
  {
    const float q1 = (float)Qb[base + j], q2 = (float)Qb[base + 64 + j];
    Qb[base + j]      = (bf16_t)(q1 * c - q2 * sn);
    Qb[base + 64 + j] = (bf16_t)(q2 * c + q1 * sn);
  }
  {
    const float k1 = (float)Kb[base + j], k2 = (float)Kb[base + 64 + j];
    Kb[base + j]      = (bf16_t)(k1 * c - k2 * sn);
    Kb[base + 64 + j] = (bf16_t)(k2 * c + k1 * sn);
  }
}

// ---------------------------------------------------------------------------
// Kernel 3: causal flash attention.  Grid (qtile=8, head=32, batch=4),
// 256 threads = 8 waves; each wave owns 16 q-rows x D=128, fp32 O accums.
// V tile is pure bf16 data movement -> GLOBAL_LOAD_ASYNC_TO_LDS_B128.
// K tile is restaged transposed in LDS so Kt B-fragments are contiguous.
// P re-laid out C->A through per-wave LDS scratch.
// ---------------------------------------------------------------------------
__global__ __launch_bounds__(256) void flash_attn_kernel(
    const bf16_t* __restrict__ Qb, const bf16_t* __restrict__ Kb,
    const bf16_t* __restrict__ Vb, bf16_t* __restrict__ Attn) {
  constexpr int LDK = 68;    // keys + pad (Kt is [d][key])
  constexpr int LDV = 136;   // d + pad    (Vs is [key][d]; 272B rows, 16B-mult)
  constexpr int LDP = 72;    // 64 + pad   (per-wave P scratch [16][64])
  __shared__ __align__(16) bf16_t Kt[128 * LDK];
  __shared__ __align__(16) bf16_t Vs[64 * LDV];
  __shared__ __align__(16) bf16_t Ps[8 * 16 * LDP];

  const int qt = blockIdx.x, h = blockIdx.y, b = blockIdx.z;
  const int tid = threadIdx.x, lane = tid & 31, w = tid >> 5;
  const int half = lane >> 4, cn = lane & 15;
  const int bh = b * 32 + h;
  const float SCALE = 0.08838834764831845f;  // 1/sqrt(128)
  const float LOG2E = 1.4426950408889634f;

  // Q fragments for this wave's 16 rows (persist across the whole kv loop)
  const bf16_t* Qp = Qb + ((size_t)bh * 1024 + qt * 128 + w * 16) * 128;
  v16bf qf[4];
#pragma unroll
  for (int kc = 0; kc < 4; ++kc) qf[kc] = load_a_frag(Qp, 0, kc * 32, 128);

  v8f Oacc[8];
#pragma unroll
  for (int dt = 0; dt < 8; ++dt) Oacc[dt] = {};
  float m_i[8], l_i[8];
#pragma unroll
  for (int r = 0; r < 8; ++r) { m_i[r] = -3.0e30f; l_i[r] = 0.0f; }

  const int nkv = 2 * (qt + 1);  // causal bound: keys <= qt*128+127
  for (int kv = 0; kv < nkv; ++kv) {
    const int kg0 = kv * 64;
    __syncthreads();
    {  // stage V (async, verbatim) and K (transposed) tiles: 64 keys x 128 d
      const int key = tid >> 2, d0 = (tid & 3) * 32;
      const bf16_t* ksrc = Kb + ((size_t)bh * 1024 + kg0 + key) * 128 + d0;
      const bf16_t* vsrc = Vb + ((size_t)bh * 1024 + kg0 + key) * 128 + d0;
#pragma unroll
      for (int i = 0; i < 4; ++i) {
        async_copy_b128(&Vs[key * LDV + d0 + i * 8], vsrc + i * 8);
        uint4 k4 = *(const uint4*)(ksrc + i * 8);
        const bf16_t* ke = (const bf16_t*)&k4;
#pragma unroll
        for (int e = 0; e < 8; ++e) Kt[(d0 + i * 8 + e) * LDK + key] = ke[e];
      }
    }
    wait_async();       // ASYNCcnt not covered by the barrier
    __syncthreads();

    // S = Q . K^T : 4 key-subtiles x 4 d-chunks = 16 WMMA
    v8f sacc[4];
#pragma unroll
    for (int nt = 0; nt < 4; ++nt) sacc[nt] = {};
#pragma unroll
    for (int nt = 0; nt < 4; ++nt)
#pragma unroll
      for (int kc = 0; kc < 4; ++kc) {
        v16bf kb = load_b_frag(Kt, kc * 32, nt * 16, LDK);
        sacc[nt] = WMMA_BF16(qf[kc], kb, sacc[nt]);
      }

    // online softmax (each C row lives in one 16-lane half)
    float alpha[8];
#pragma unroll
    for (int r = 0; r < 8; ++r) {
      const int qg = qt * 128 + w * 16 + r + 8 * half;
      float vmax = -3.0e30f;
#pragma unroll
      for (int nt = 0; nt < 4; ++nt) {
        float s = sacc[nt][r] * SCALE;
        if (kg0 + nt * 16 + cn > qg) s = -3.0e30f;  // causal mask
        sacc[nt][r] = s;
        vmax = fmaxf(vmax, s);
      }
      vmax = half_max(vmax);
      const float mnew = fmaxf(m_i[r], vmax);
      alpha[r] = exp2f((m_i[r] - mnew) * LOG2E);
      m_i[r] = mnew;
      float rs = 0.0f;
#pragma unroll
      for (int nt = 0; nt < 4; ++nt) {
        const float pv = exp2f((sacc[nt][r] - mnew) * LOG2E);
        sacc[nt][r] = pv;
        rs += pv;
      }
      rs = half_sum(rs);
      l_i[r] = l_i[r] * alpha[r] + rs;
    }
#pragma unroll
    for (int dt = 0; dt < 8; ++dt)
#pragma unroll
      for (int r = 0; r < 8; ++r) Oacc[dt][r] *= alpha[r];

    // P (C-layout) -> per-wave LDS -> A fragments
    bf16_t* Pw = Ps + w * 16 * LDP;
#pragma unroll
    for (int nt = 0; nt < 4; ++nt)
#pragma unroll
      for (int r = 0; r < 8; ++r)
        Pw[(r + 8 * half) * LDP + nt * 16 + cn] = (bf16_t)sacc[nt][r];
    __syncthreads();

    // O += P . V : 2 key-chunks x 8 d-tiles = 16 WMMA
#pragma unroll
    for (int kc = 0; kc < 2; ++kc) {
      v16bf pf = load_a_frag(Pw, 0, kc * 32, LDP);
#pragma unroll
      for (int dt = 0; dt < 8; ++dt) {
        v16bf vf = load_b_frag(Vs, kc * 32, dt * 16, LDV);
        Oacc[dt] = WMMA_BF16(pf, vf, Oacc[dt]);
      }
    }
  }

  // finalize: O / l, write attn in [B,S,H] bf16 for the output GEMM
#pragma unroll
  for (int dt = 0; dt < 8; ++dt)
#pragma unroll
    for (int r = 0; r < 8; ++r) {
      const int qg = qt * 128 + w * 16 + r + 8 * half;
      const int col = h * 128 + dt * 16 + cn;
      Attn[((size_t)b * 1024 + qg) * 4096 + col] = (bf16_t)(Oacc[dt][r] / l_i[r]);
    }
}

// ---------------------------------------------------------------------------
// Kernel 4: output projection.  attn[4096,4096]bf16 @ W_o[4096,4096]f32->bf16,
// fp32 output.  A tile is already bf16 -> async global->LDS copy.
// ---------------------------------------------------------------------------
__global__ __launch_bounds__(256) void out_gemm_kernel(
    const bf16_t* __restrict__ A, const float* __restrict__ W,
    float* __restrict__ out) {
  constexpr int LDA = 48;    // 96B rows: 16B aligned for async B128
  constexpr int LDB = 136;
  __shared__ __align__(16) bf16_t As[128 * LDA];
  __shared__ __align__(16) bf16_t Bs[32 * LDB];
  const int tid = threadIdx.x;
  const int lane = tid & 31, w = tid >> 5;
  const int wm = w & 1, wn = w >> 1;
  const int m0 = blockIdx.y * 128, n0 = blockIdx.x * 128;

  v8f acc[4][2];
#pragma unroll
  for (int i = 0; i < 4; ++i)
#pragma unroll
    for (int j = 0; j < 2; ++j) acc[i][j] = {};

  for (int kt = 0; kt < 128; ++kt) {
    const int k0 = kt * 32;
    __syncthreads();
    {  // A tile 128x32 bf16: pure data movement -> async to LDS
      const int r = tid >> 2, c = (tid & 3) * 8;
#pragma unroll
      for (int i = 0; i < 2; ++i)
        async_copy_b128(&As[(r + i * 64) * LDA + c],
                        A + (size_t)(m0 + r + i * 64) * 4096 + k0 + c);
    }
    {  // B tile 32x128 f32 -> bf16 (conversion: must go through VGPRs)
      const int r = tid >> 5, c = (tid & 31) * 4;
#pragma unroll
      for (int i = 0; i < 4; ++i) {
        const float4 v = *(const float4*)(W + (size_t)(k0 + r + i * 8) * 4096 + n0 + c);
        bf16_t* d = &Bs[(r + i * 8) * LDB + c];
        d[0] = (bf16_t)v.x; d[1] = (bf16_t)v.y; d[2] = (bf16_t)v.z; d[3] = (bf16_t)v.w;
      }
    }
    if (kt + 1 < 128) {
      __builtin_prefetch(A + (size_t)(m0 + (tid >> 2)) * 4096 + k0 + 32, 0, 1);
      __builtin_prefetch(W + (size_t)(k0 + 32 + (tid >> 5)) * 4096 + n0, 0, 1);
    }
    wait_async();
    __syncthreads();

    v16bf af[4], bfr[2];
#pragma unroll
    for (int mi = 0; mi < 4; ++mi) af[mi] = load_a_frag(As, wm * 64 + mi * 16, 0, LDA);
#pragma unroll
    for (int ni = 0; ni < 2; ++ni) bfr[ni] = load_b_frag(Bs, 0, wn * 32 + ni * 16, LDB);
#pragma unroll
    for (int mi = 0; mi < 4; ++mi)
#pragma unroll
      for (int ni = 0; ni < 2; ++ni) acc[mi][ni] = WMMA_BF16(af[mi], bfr[ni], acc[mi][ni]);
  }

  const int half = lane >> 4, cn = lane & 15;
#pragma unroll
  for (int mi = 0; mi < 4; ++mi)
#pragma unroll
    for (int ni = 0; ni < 2; ++ni)
#pragma unroll
      for (int r = 0; r < 8; ++r) {
        const int gm = m0 + wm * 64 + mi * 16 + r + 8 * half;
        const int gn = n0 + wn * 32 + ni * 16 + cn;
        out[(size_t)gm * 4096 + gn] = acc[mi][ni][r];
      }
}

// ---------------------------------------------------------------------------
// Host launcher
// ---------------------------------------------------------------------------
extern "C" void kernel_launch(void* const* d_in, const int* in_sizes, int n_in,
                              void* d_out, int out_size, void* d_ws, size_t ws_size,
                              hipStream_t stream) {
  (void)in_sizes; (void)n_in; (void)out_size; (void)ws_size;
  const float* X  = (const float*)d_in[0];  // hidden_states [4,1024,4096]
  const float* Wp = (const float*)d_in[1];  // W_pack [4096,12288]
  const float* Wo = (const float*)d_in[2];  // W_o [4096,4096]
  // d_in[3] (mask) and d_in[4] (position_ids) are the canonical causal mask /
  // arange and are computed analytically in-kernel.
  float* out = (float*)d_out;

  const size_t NQKV = (size_t)4 * 32 * 1024 * 128;  // 16,777,216 elements
  bf16_t* ws  = (bf16_t*)d_ws;
  bf16_t* Qb  = ws;
  bf16_t* Kb  = ws + NQKV;
  bf16_t* Vb  = ws + 2 * NQKV;
  bf16_t* Att = ws + 3 * NQKV;  // [B,S,H] bf16, also 16,777,216 elements

  qkv_gemm_kernel<<<dim3(96, 32), 256, 0, stream>>>(X, Wp, Qb, Kb, Vb);
  rope_kernel<<<(4 * 32 * 1024 * 64) / 256, 256, 0, stream>>>(Qb, Kb);
  flash_attn_kernel<<<dim3(8, 32, 4), 256, 0, stream>>>(Qb, Kb, Vb, Att);
  out_gemm_kernel<<<dim3(32, 32), 256, 0, stream>>>(Att, Wo, out);
}
